// TriplaneAttention_15144054686221
// MI455X (gfx1250) — compile-verified, matrix operand order checked by reading, not run on hardware
//
#include <hip/hip_runtime.h>
#include <hip/hip_bf16.h>

// Types for WMMA operands (CDNA5 wave32: A/B = v16h, C/D = v8f for 16x16x32 f16)
typedef _Float16 h2  __attribute__((ext_vector_type(2)));
typedef _Float16 h4  __attribute__((ext_vector_type(4)));
typedef _Float16 h8  __attribute__((ext_vector_type(8)));
typedef _Float16 h16 __attribute__((ext_vector_type(16)));
typedef float    f8  __attribute__((ext_vector_type(8)));
typedef float    f4  __attribute__((ext_vector_type(4)));
typedef unsigned u32x4 __attribute__((ext_vector_type(4)));
typedef int      i32x4 __attribute__((ext_vector_type(4)));
typedef int      i32x8 __attribute__((ext_vector_type(8)));

#define SITES 8192   // B*HWC = 2*4096
#define CDIM  512
#define NH    8
#define DH    64
#define LCT   32

#if __has_builtin(__builtin_amdgcn_tensor_load_to_lds) && \
    __has_builtin(__builtin_amdgcn_s_wait_tensorcnt)
#define HAVE_TDM 1
#endif

#define WMMA(A, B, C) __builtin_amdgcn_wmma_f32_16x16x32_f16(false, (A), false, (B), (short)0, (C), false, false)

// A / B^T fragment for v_wmma_f32_16x16x32_f16 from row-major f16 storage
// (LDS or global). ISA 7.12.2 16-bit A 16x32 layout: lane 0-15 row M=lane,
// halves = [k0..k0+7] then [k0+16..k0+23]; lanes 16-31 add +8 to k base.
static __device__ __forceinline__ h16 frag_f16(const _Float16* base, int row,
                                               int stride, int k0, int lane) {
    const _Float16* p = base + (size_t)row * stride + k0 + ((lane >> 4) << 3);
    h8 lo = *(const h8*)p;
    h8 hi = *(const h8*)(p + 16);
    return __builtin_shufflevector(lo, hi, 0, 1, 2, 3, 4, 5, 6, 7,
                                           8, 9, 10, 11, 12, 13, 14, 15);
}

// Same fragment but sourced from a row-major f32 row (convert on the fly).
static __device__ __forceinline__ h16 frag_f32row(const float* rowp, int k0, int lane) {
    int kb = k0 + ((lane >> 4) << 3);
    f4 a0 = *(const f4*)(rowp + kb);
    f4 a1 = *(const f4*)(rowp + kb + 4);
    f4 b0 = *(const f4*)(rowp + kb + 16);
    f4 b1 = *(const f4*)(rowp + kb + 20);
    h16 r;
    r[0]  = (_Float16)a0[0]; r[1]  = (_Float16)a0[1]; r[2]  = (_Float16)a0[2]; r[3]  = (_Float16)a0[3];
    r[4]  = (_Float16)a1[0]; r[5]  = (_Float16)a1[1]; r[6]  = (_Float16)a1[2]; r[7]  = (_Float16)a1[3];
    r[8]  = (_Float16)b0[0]; r[9]  = (_Float16)b0[1]; r[10] = (_Float16)b0[2]; r[11] = (_Float16)b0[3];
    r[12] = (_Float16)b1[0]; r[13] = (_Float16)b1[1]; r[14] = (_Float16)b1[2]; r[15] = (_Float16)b1[3];
    return r;
}

// ---------------- Kernel 0: LayerNorm(query) -> qln (f16) ----------------
__global__ __launch_bounds__(256) void ln_kernel(const float* __restrict__ q,
                                                 const float* __restrict__ g,
                                                 const float* __restrict__ b,
                                                 _Float16* __restrict__ qln) {
    int wave = threadIdx.x >> 5, lane = threadIdx.x & 31;
    int site = blockIdx.x * 8 + wave;     // one site per wave32
    const float* row = q + (size_t)site * CDIM;
    f4 x[4];
    float s = 0.f, s2 = 0.f;
    for (int i = 0; i < 4; ++i) {
        f4 v = *(const f4*)(row + lane * 4 + 128 * i);
        x[i] = v;
        s  += v[0] + v[1] + v[2] + v[3];
        s2 += v[0]*v[0] + v[1]*v[1] + v[2]*v[2] + v[3]*v[3];
    }
    for (int o = 16; o; o >>= 1) { s += __shfl_xor(s, o); s2 += __shfl_xor(s2, o); }
    float mu   = s * (1.f / CDIM);
    float var  = s2 * (1.f / CDIM) - mu * mu;
    float rstd = rsqrtf(var + 1e-5f);
    _Float16* orow = qln + (size_t)site * CDIM;
    for (int i = 0; i < 4; ++i) {
        int c = lane * 4 + 128 * i;
        h4 hv;
        for (int j = 0; j < 4; ++j)
            hv[j] = (_Float16)((x[i][j] - mu) * rstd * g[c + j] + b[c + j]);
        *(h4*)(orow + c) = hv;
    }
}

// ---- Kernel 1: qk[s,h,c] = sum_d qln[s,h*64+d] * Wk[h*64+d,c]  (k_b dropped:
// ---- constant over l, softmax is shift-invariant) ----
__global__ __launch_bounds__(256) void qk_kernel(const _Float16* __restrict__ qln,
                                                 const float* __restrict__ kw,
                                                 _Float16* __restrict__ qk) {
    __shared__ _Float16 Bt[CDIM * 72];            // Bt[c][d], padded stride
    int h  = blockIdx.y;
    int s0 = blockIdx.x * 16;
    int lane = threadIdx.x & 31, wave = threadIdx.x >> 5;

    const float* wbase = kw + (size_t)(h * DH) * CDIM;   // Wk rows of this head
    for (int idx = threadIdx.x; idx < DH * CDIM / 4; idx += 256) {
        int d = idx >> 7, c4 = (idx & 127) << 2;
        f4 w = *(const f4*)(wbase + (size_t)d * CDIM + c4);
        Bt[(c4 + 0) * 72 + d] = (_Float16)w[0];
        Bt[(c4 + 1) * 72 + d] = (_Float16)w[1];
        Bt[(c4 + 2) * 72 + d] = (_Float16)w[2];
        Bt[(c4 + 3) * 72 + d] = (_Float16)w[3];
    }
    __syncthreads();

    // A = qln[s0..s0+15, h*64 .. h*64+63]  (two K=32 steps), reused for all N tiles
    const _Float16* abase = qln + (size_t)s0 * CDIM + h * DH;
    h16 a0 = frag_f16(abase, lane & 15, CDIM, 0,  lane);
    h16 a1 = frag_f16(abase, lane & 15, CDIM, 32, lane);

    for (int t = wave; t < 32; t += 8) {          // 32 N tiles over c, 8 waves
        int c0 = t * 16;
        h16 b0 = frag_f16(Bt, c0 + (lane & 15), 72, 0,  lane);
        h16 b1 = frag_f16(Bt, c0 + (lane & 15), 72, 32, lane);
        f8 acc = {};
        acc = WMMA(a0, b0, acc);
        acc = WMMA(a1, b1, acc);
        int c  = c0 + (lane & 15);
        int mb = (lane >> 4) << 3;
        for (int i = 0; i < 8; ++i) {
            int s = s0 + mb + i;
            qk[((size_t)s * NH + h) * CDIM + c] = (_Float16)acc[i];
        }
    }
}

// ---- Kernel 2: per-site scores -> softmax -> agg (streams context ONCE,
// ---- non-temporal: context (512MB) must not thrash the 192MB L2) ----
__global__ __launch_bounds__(256) void attn_kernel(const float* __restrict__ ctx,
                                                   const _Float16* __restrict__ qk,
                                                   _Float16* __restrict__ agg) {
    __shared__ _Float16 ctxh[LCT * 520];    // ctx[l][c]   (scores B^T layout)
    __shared__ _Float16 ctxT[CDIM * 40];    // ctx[c][l]   (agg    B^T layout)
    __shared__ _Float16 qkh[16 * 520];      // qk[h][c], rows 8..15 zero (M pad)
    __shared__ _Float16 aw[16 * 40];        // softmax weights [h][l], rows 8..15 zero
    __shared__ float    sc[8 * 16 * 32];    // per-wave partial score tiles

    int s = blockIdx.x;
    int lane = threadIdx.x & 31, wave = threadIdx.x >> 5;
    const _Float16* qb = qk + (size_t)s * NH * CDIM;

#ifdef HAVE_TDM
    // Tensor Data Mover: DMA the 8x512 f16 qk tile into LDS with a 16B pad per
    // 1KB row (row stride 512 -> 520 halves). Issued by wave 0 only (TDM done
    // signal is per instruction); fenced with s_wait_tensorcnt before barrier.
    if (threadIdx.x < 32) {
        unsigned long long ga = (unsigned long long)(uintptr_t)qb;
        u32x4 g0;
        g0[0] = 1u;                                   // count=1, user descriptor
        g0[1] = (unsigned)(uintptr_t)&qkh[0];         // lds_addr (bytes)
        g0[2] = (unsigned)ga;                         // global_addr[31:0]
        g0[3] = (unsigned)((ga >> 32) & 0x01ffffffu) | (2u << 30); // addr[56:32]|type=2
        i32x8 g1 = {};
        g1[0] = (1 << 16)      // data_size = 2 bytes
              | (1 << 20)      // pad_enable
              | (7 << 22)      // pad_interval: 256 DWORDs (= one 1KB row)
              | (3 << 25);     // pad_amount: 4 DWORDs (= 16B = 8 halves)
        g1[1] = (int)(512u << 16);                    // tensor_dim0 = 512
        g1[2] = (int)(8u << 16);                      // tensor_dim1 = 8
        g1[3] = (int)(512u << 16);                    // tile_dim0 = 512
        g1[4] = 8;                                    // tile_dim1 = 8
        g1[5] = 512;                                  // tensor_dim0_stride = 512
        i32x4 z4 = {};
#if defined(__clang_major__) && (__clang_major__ >= 23)
        i32x8 z8 = {};
        __builtin_amdgcn_tensor_load_to_lds(g0, g1, z4, z4, z8, 0);
#else
        __builtin_amdgcn_tensor_load_to_lds(g0, g1, z4, z4, 0);
#endif
        __builtin_amdgcn_s_wait_tensorcnt(0);
    }
#endif

    // Stage context: f32x4 NT loads, two l-rows per iter so transposed copy can
    // use packed b32 stores.
    const float* cb = ctx + (size_t)s * LCT * CDIM;
    for (int idx = threadIdx.x; idx < (LCT / 2) * (CDIM / 4); idx += 256) {
        int l  = (idx >> 7) << 1;            // 0,2,..,30
        int c4 = (idx & 127) << 2;
        f4 v0 = __builtin_nontemporal_load((const f4*)(cb + (size_t)l * CDIM + c4));
        f4 v1 = __builtin_nontemporal_load((const f4*)(cb + (size_t)(l + 1) * CDIM + c4));
        h4 hv0, hv1;
        for (int j = 0; j < 4; ++j) { hv0[j] = (_Float16)v0[j]; hv1[j] = (_Float16)v1[j]; }
        *(h4*)&ctxh[l * 520 + c4]       = hv0;
        *(h4*)&ctxh[(l + 1) * 520 + c4] = hv1;
        for (int j = 0; j < 4; ++j) {
            h2 p; p[0] = hv0[j]; p[1] = hv1[j];
            *(h2*)&ctxT[(c4 + j) * 40 + l] = p;
        }
    }
#ifndef HAVE_TDM
    // Fallback: stage qk rows with b128 copies.
    for (int idx = threadIdx.x; idx < NH * CDIM / 8; idx += 256) {
        int hh = idx >> 6, c8 = (idx & 63) << 3;
        *(h8*)&qkh[hh * 520 + c8] = *(const h8*)(qb + ((size_t)idx << 3));
    }
#endif
    for (int idx = threadIdx.x; idx < 8 * 520 / 8; idx += 256) {
        h8 z = {};
        *(h8*)&qkh[8 * 520 + idx * 8] = z;            // zero M-padding rows 8..15
    }
    for (int idx = threadIdx.x; idx < 16 * 40 / 8; idx += 256) {
        h8 z = {};
        *(h8*)&aw[idx * 8] = z;
    }
    __syncthreads();

    // scores[h,l] = sum_c qk[h,c]*ctx[l,c] ; K=512 split 8 ways across waves
    f8 acc0 = {}, acc1 = {};
    for (int j = 0; j < 2; ++j) {
        int k0 = (wave * 2 + j) * 32;
        h16 a  = frag_f16(qkh,  lane & 15,        520, k0, lane);
        h16 b0 = frag_f16(ctxh, (lane & 15),      520, k0, lane);  // l = 0..15
        h16 b1 = frag_f16(ctxh, 16 + (lane & 15), 520, k0, lane);  // l = 16..31
        acc0 = WMMA(a, b0, acc0);
        acc1 = WMMA(a, b1, acc1);
    }
    {
        int n = lane & 15, mb = (lane >> 4) << 3;
        for (int i = 0; i < 8; ++i) {
            sc[(wave * 16 + mb + i) * 32 + n]      = acc0[i];
            sc[(wave * 16 + mb + i) * 32 + 16 + n] = acc1[i];
        }
    }
    __syncthreads();

    // cross-wave reduce + softmax over L=32 (no 1/sqrt(d) scale, per reference)
    if (threadIdx.x < NH) {
        int h = threadIdx.x;
        float v[32], mx = -1e30f;
        for (int l = 0; l < 32; ++l) {
            float t = 0.f;
            for (int w = 0; w < 8; ++w) t += sc[(w * 16 + h) * 32 + l];
            v[l] = t; mx = fmaxf(mx, t);
        }
        float sum = 0.f;
        for (int l = 0; l < 32; ++l) { float e = __expf(v[l] - mx); v[l] = e; sum += e; }
        float inv = 1.f / sum;
        for (int l = 0; l < 32; ++l) aw[h * 40 + l] = (_Float16)(v[l] * inv);
    }
    __syncthreads();

    // agg[h,c] = sum_l w[h,l]*ctx[l,c] ; K=32 == L, exactly one WMMA K-step
    h16 a = frag_f16(aw, lane & 15, 40, 0, lane);
    for (int t = wave; t < 32; t += 8) {
        int c0 = t * 16;
        h16 b = frag_f16(ctxT, c0 + (lane & 15), 40, 0, lane);
        f8 z = {};
        f8 acc = WMMA(a, b, z);
        if (lane < 16) {                          // rows m=0..7 only (valid heads)
            int c = c0 + lane;
            for (int i = 0; i < 8; ++i)
                agg[((size_t)s * NH + i) * CDIM + c] = (_Float16)acc[i];
        }
    }
}

// ---- Kernel 3: out[s,h,d] = sum_c agg[s,h,c]*Wv[h*64+d,c] + v_b, with the
// ---- reference's faithful transpose/reshape scatter ----
__global__ __launch_bounds__(128) void out_kernel(const _Float16* __restrict__ agg,
                                                  const float* __restrict__ vw,
                                                  const float* __restrict__ vb,
                                                  float* __restrict__ out) {
    int h = blockIdx.y, s0 = blockIdx.x * 16;
    int lane = threadIdx.x & 31, wave = threadIdx.x >> 5;
    int d0 = wave * 16;                                   // N tile over d (4 waves)
    const float*     wrow = vw + (size_t)(h * DH + d0 + (lane & 15)) * CDIM;
    const _Float16*  ab   = agg + ((size_t)s0 * NH + h) * CDIM;

    f8 acc = {};
    for (int kk = 0; kk < 16; ++kk) {
        int k0 = kk * 32;
        h16 a = frag_f16(ab, lane & 15, NH * CDIM, k0, lane);
        h16 b = frag_f32row(wrow, k0, lane);
        acc = WMMA(a, b, acc);
    }
    int d = d0 + (lane & 15);
    float bias = vb[h * DH + d];
    int mb = (lane >> 4) << 3;
    for (int i = 0; i < 8; ++i) {
        int s = s0 + mb + i;
        int bidx = s >> 12, nimg = s & 4095;
        int rowp = h * 512 + (nimg >> 3);                 // transpose(1,2)+reshape map
        int colp = ((nimg & 7) << 6) + d;
        __builtin_nontemporal_store(acc[i] + bias,
            &out[((size_t)bidx * 4096 + rowp) * 512 + colp]);
    }
}

extern "C" void kernel_launch(void* const* d_in, const int* in_sizes, int n_in,
                              void* d_out, int out_size, void* d_ws, size_t ws_size,
                              hipStream_t stream) {
    const float* query   = (const float*)d_in[0];
    const float* context = (const float*)d_in[1];
    const float* k_w     = (const float*)d_in[2];
    // d_in[3] = k_b: mathematically dropped (constant over L -> softmax invariant)
    const float* v_w     = (const float*)d_in[4];
    const float* v_b     = (const float*)d_in[5];
    const float* ln_g    = (const float*)d_in[6];
    const float* ln_b    = (const float*)d_in[7];
    float* out = (float*)d_out;

    char* ws = (char*)d_ws;
    _Float16* qln = (_Float16*)ws;                                 //  8 MiB
    _Float16* qk  = (_Float16*)(ws + (size_t)8  * 1024 * 1024);    // 64 MiB
    _Float16* agg = (_Float16*)(ws + (size_t)72 * 1024 * 1024);    // 64 MiB

    ln_kernel  <<<SITES / 8, 256, 0, stream>>>(query, ln_g, ln_b, qln);
    qk_kernel  <<<dim3(SITES / 16, NH), 256, 0, stream>>>(qln, k_w, qk);
    attn_kernel<<<SITES, 256, 0, stream>>>(context, qk, agg);
    out_kernel <<<dim3(SITES / 16, NH), 128, 0, stream>>>(agg, v_w, v_b, out);
}
